// PointMVS_Transformer_54443005444382
// MI455X (gfx1250) — compile-verified, matrix operand order checked by reading, not run on hardware
//
#include <hip/hip_runtime.h>
#include <hip/hip_bf16.h>
#include <math.h>

// ---------------------------------------------------------------------------
// PointMVS transformer pipeline for MI455X (gfx1250, wave32, WMMA).
// Shapes hard-coded from setup_inputs(): B=2, C=64, N=32768, k=16, window=17.
// ---------------------------------------------------------------------------

typedef __attribute__((ext_vector_type(16))) _Float16 v16h;
typedef __attribute__((ext_vector_type(8)))  float    v8f;

#define BB   2
#define CC   64
#define NN   32768
#define KNN  16
#define WLEN 16
#define WIN  17
#define BN   (BB * NN)               // 65536
#define FSZ  ((size_t)BB * CC * NN)  // 4,194,304 floats per (B,C,N) tensor

// ---------------------------------------------------------------------------
// Kernel 1: per-point sim against 16 kNN neighbors; emit index of the
// SECOND-largest similarity (softmax is monotone -> argsort(sim)==argsort(attn);
// the walk only ever uses top4[...,1]).
// ---------------------------------------------------------------------------
__global__ __launch_bounds__(256) void walk_top4_kernel(
    const float* __restrict__ feature, const int* __restrict__ knn,
    int* __restrict__ next1)
{
  int p = blockIdx.x * 256 + threadIdx.x;      // flat point id in [0, B*N)
  if (p >= BN) return;
  int b = p >> 15;                              // p / N
  int n = p & (NN - 1);
  const float* fb = feature + (size_t)b * CC * NN;

  int idx[KNN];
#pragma unroll
  for (int k = 0; k < KNN; ++k) idx[k] = knn[(size_t)p * KNN + k];

  float sim[KNN];
#pragma unroll
  for (int k = 0; k < KNN; ++k) sim[k] = 0.0f;

  for (int c = 0; c < CC; ++c) {
    float fc = fb[(size_t)c * NN + n];
#pragma unroll
    for (int k = 0; k < KNN; ++k)
      sim[k] = fmaf(fc, fb[(size_t)c * NN + idx[k]], sim[k]);
  }

  // argmax (ties -> lowest index, matching lax.top_k), then 2nd argmax.
  int i1 = 0; float v1 = sim[0];
#pragma unroll
  for (int k = 1; k < KNN; ++k) if (sim[k] > v1) { v1 = sim[k]; i1 = k; }
  int i2 = (i1 == 0) ? 1 : 0; float v2 = sim[i2];
#pragma unroll
  for (int k = 0; k < KNN; ++k)
    if (k != i1 && sim[k] > v2) { v2 = sim[k]; i2 = k; }

  next1[p] = i2;
}

// ---------------------------------------------------------------------------
// Kernel 2: walk recurrence. Reproduces the reference's flattened re-indexing:
// nxt = next1[p]; then 15x: nxt = next1[nxt]  (nxt in [0,16)).
// ---------------------------------------------------------------------------
__global__ __launch_bounds__(256) void walk_build_kernel(
    const int* __restrict__ next1, int* __restrict__ walk)
{
  int p = blockIdx.x * 256 + threadIdx.x;
  if (p >= BN) return;
  int v = next1[p];
  walk[(size_t)p * WLEN + 0] = v;
#pragma unroll
  for (int j = 1; j < WLEN; ++j) {
    v = next1[v];                 // v in [0,16): rows 0..15 of flattened array
    walk[(size_t)p * WLEN + j] = v;
  }
}

// ---------------------------------------------------------------------------
// Kernel 3: QKV projection as a dense GEMM:  O(64, N) = W(64,64) @ f(64, N) + b
// One wave computes one 16-column tile for all 64 rows of Q, K and V.
// v_wmma_f32_16x16x32_f16, two K-steps (K=64). No divergence: EXEC all-1s.
// The per-matrix work is an inlined helper taking DIRECT kernel-arg pointers,
// so clang keeps addrspace(1) inference -> global_load/global_store (flat ops
// would couple LOADcnt with DScnt and serialize on s_wait_loadcnt_dscnt).
// ---------------------------------------------------------------------------
__device__ __forceinline__ void qkv_one_matrix(
    const float* __restrict__ W, const float* __restrict__ bias,
    float* __restrict__ O, const v16h& B0, const v16h& B1,
    int mrow, int koff, int lanehi8, int col)
{
#pragma unroll
  for (int mt = 0; mt < 4; ++mt) {
    const float* __restrict__ wr = W + (size_t)(mt * 16 + mrow) * CC;
    v16h A0, A1;
#pragma unroll
    for (int h = 0; h < 8; ++h) {
      A0[h]     = (_Float16)wr[ 0 + koff + h];
      A0[h + 8] = (_Float16)wr[16 + koff + h];
      A1[h]     = (_Float16)wr[32 + koff + h];
      A1[h + 8] = (_Float16)wr[48 + koff + h];
    }
    v8f acc = {};
    acc = __builtin_amdgcn_wmma_f32_16x16x32_f16(
              false, A0, false, B0, (short)0, acc, false, false);
    acc = __builtin_amdgcn_wmma_f32_16x16x32_f16(
              false, A1, false, B1, (short)0, acc, false, false);

    // D layout: VGPR r holds M = mbase + r, N = col (lane&15).
    const int mbase = mt * 16 + lanehi8;
    // Preload the 8 bias values with two b128 loads (16B-aligned).
    const float4 bv0 = *(const float4*)(bias + mbase);
    const float4 bv1 = *(const float4*)(bias + mbase + 4);
    float bb[8] = {bv0.x, bv0.y, bv0.z, bv0.w, bv1.x, bv1.y, bv1.z, bv1.w};
    // Single 64-bit base; row stores become immediate-offset global stores.
    float* __restrict__ ob = O + (size_t)mbase * NN + col;
#pragma unroll
    for (int r = 0; r < 8; ++r)
      ob[(size_t)r * NN] = acc[r] + bb[r];
  }
}

__global__ __launch_bounds__(256) void gemm_qkv_kernel(
    const float* __restrict__ f,
    const float* __restrict__ Wq, const float* __restrict__ bq,
    const float* __restrict__ Wk, const float* __restrict__ bk,
    const float* __restrict__ Wv, const float* __restrict__ bv,
    float* __restrict__ Qf, float* __restrict__ Kf, float* __restrict__ Vf)
{
  const int wave = threadIdx.x >> 5;
  const int lane = threadIdx.x & 31;
  const int tile = blockIdx.x * 8 + wave;       // n-tile in [0, 2048)
  const int b    = blockIdx.y;
  const float* __restrict__ fb = f + (size_t)b * CC * NN;
  const int col    = tile * 16 + (lane & 15);
  const int koff   = (lane >> 4) * 8;           // upper half-wave owns K+8 block
  const int lanehi = (lane >> 4) << 3;

  // Prefetch next tile's feature columns into GL2 (global_prefetch_b8).
  __builtin_prefetch(fb + (size_t)koff * NN + col + 128, 0, 1);

  // B operand tiles: kstep0 covers K=0..31, kstep1 covers K=32..63.
  const float* __restrict__ fcol = fb + col;
  v16h B0, B1;
#pragma unroll
  for (int h = 0; h < 8; ++h) {
    B0[h]     = (_Float16)fcol[(size_t)( 0 + koff + h) * NN];
    B0[h + 8] = (_Float16)fcol[(size_t)(16 + koff + h) * NN];
    B1[h]     = (_Float16)fcol[(size_t)(32 + koff + h) * NN];
    B1[h + 8] = (_Float16)fcol[(size_t)(48 + koff + h) * NN];
  }

  const size_t bo = (size_t)b * CC * NN;
  qkv_one_matrix(Wq, bq, Qf + bo, B0, B1, lane & 15, koff, lanehi, col);
  qkv_one_matrix(Wk, bk, Kf + bo, B0, B1, lane & 15, koff, lanehi, col);
  qkv_one_matrix(Wv, bv, Vf + bo, B0, B1, lane & 15, koff, lanehi, col);
}

// ---------------------------------------------------------------------------
// Kernel 4: per-point window attention. One wave per point, Q/K/V windows
// staged in LDS; 17x17 sim; only the softmax diagonal is needed; out = d.V.
// Channel mean/var partials reduced via ds_add_f32 then one global atomic
// per channel per block.
// ---------------------------------------------------------------------------
__global__ __launch_bounds__(128) void attn_kernel(
    const float* __restrict__ Qf, const float* __restrict__ Kf,
    const float* __restrict__ Vf, const int* __restrict__ walk,
    float* __restrict__ xout, float* __restrict__ gsum, float* __restrict__ gsq)
{
  __shared__ float Qw[4][WIN][CC];
  __shared__ float Kw[4][WIN][CC];
  __shared__ float Vw[4][WIN][CC];
  __shared__ float S [4][WIN][WIN + 1];
  __shared__ float dv[4][WIN];
  __shared__ float chsum[CC], chsq[CC];

  const int wave = threadIdx.x >> 5;
  const int lane = threadIdx.x & 31;
  const int p = blockIdx.x * 4 + wave;          // [0, 65536)
  const int b = p >> 15;
  const int n = p & (NN - 1);
  const float* __restrict__ Qb = Qf + (size_t)b * CC * NN;
  const float* __restrict__ Kb = Kf + (size_t)b * CC * NN;
  const float* __restrict__ Vb = Vf + (size_t)b * CC * NN;

  if (threadIdx.x < CC) { chsum[threadIdx.x] = 0.0f; chsq[threadIdx.x] = 0.0f; }
  __syncthreads();

  // Gather the 17 window columns (16 walk entries + center last).
  for (int i = 0; i < WIN; ++i) {
    const int coli = (i < WLEN) ? walk[(size_t)p * WLEN + i] : n;
    for (int c = lane; c < CC; c += 32) {
      Qw[wave][i][c] = Qb[(size_t)c * NN + coli];
      Kw[wave][i][c] = Kb[(size_t)c * NN + coli];
      Vw[wave][i][c] = Vb[(size_t)c * NN + coli];
    }
  }
  __syncthreads();

  // sim[i][j] = Q_i . K_j
  for (int t = lane; t < WIN * WIN; t += 32) {
    const int i = t / WIN, j = t % WIN;
    float s = 0.0f;
#pragma unroll 8
    for (int c = 0; c < CC; ++c) s = fmaf(Qw[wave][i][c], Kw[wave][j][c], s);
    S[wave][i][j] = s;
  }
  __syncthreads();

  // Diagonal of row-softmax.
  if (lane < WIN) {
    const int i = lane;
    float mx = -3.4e38f;
    for (int j = 0; j < WIN; ++j) mx = fmaxf(mx, S[wave][i][j]);
    float sum = 0.0f;
    for (int j = 0; j < WIN; ++j) sum += __expf(S[wave][i][j] - mx);
    dv[wave][i] = __expf(S[wave][i][i] - mx) / sum;
  }
  __syncthreads();

  // out[c] = sum_i d_i * V[i][c]; accumulate channel stats in LDS.
  for (int c = lane; c < CC; c += 32) {
    float o = 0.0f;
#pragma unroll
    for (int i = 0; i < WIN; ++i) o = fmaf(dv[wave][i], Vw[wave][i][c], o);
    xout[((size_t)b * CC + c) * NN + n] = o;
    atomicAdd(&chsum[c], o);
    atomicAdd(&chsq[c], o * o);
  }
  __syncthreads();

  if (threadIdx.x < CC) {
    atomicAdd(&gsum[threadIdx.x], chsum[threadIdx.x]);
    atomicAdd(&gsq [threadIdx.x], chsq [threadIdx.x]);
  }
}

// ---------------------------------------------------------------------------
// Kernel 5: channel norm + ReLU + residual; updates f_cur in place and writes
// this layer's slice of the (B, 192, N) output.
// ---------------------------------------------------------------------------
__global__ __launch_bounds__(256) void norm_kernel(
    const float* __restrict__ xout, float* __restrict__ fcur,
    const float* __restrict__ gamma, const float* __restrict__ beta,
    const float* __restrict__ gsum, const float* __restrict__ gsq,
    float* __restrict__ out, int layer)
{
  size_t t = (size_t)blockIdx.x * 256 + threadIdx.x;  // [0, B*C*N)
  if (t >= FSZ) return;
  const int n  = (int)(t & (NN - 1));
  const int bc = (int)(t >> 15);
  const int c  = bc & (CC - 1);
  const int b  = bc >> 6;

  const float inv  = 1.0f / (float)BN;
  const float mean = gsum[c] * inv;
  const float var  = gsq[c] * inv - mean * mean;
  float y = gamma[c] * (xout[t] - mean) * rsqrtf(var + 1e-5f) + beta[c];
  y = fmaxf(y, 0.0f);
  const float fn = fcur[t] + y;
  fcur[t] = fn;
  out[(((size_t)b * 3 * CC) + (size_t)layer * CC + c) * NN + n] = fn;
}

__global__ void zero_kernel(float* g) { if (threadIdx.x < 128) g[threadIdx.x] = 0.0f; }

// ---------------------------------------------------------------------------
extern "C" void kernel_launch(void* const* d_in, const int* in_sizes, int n_in,
                              void* d_out, int out_size, void* d_ws, size_t ws_size,
                              hipStream_t stream) {
  (void)in_sizes; (void)n_in; (void)out_size; (void)ws_size;
  const float* feature = (const float*)d_in[0];
  const int*   knn     = (const int*)  d_in[1];
  const float* Wq = (const float*)d_in[2];
  const float* bq = (const float*)d_in[3];
  const float* Wk = (const float*)d_in[4];
  const float* bk = (const float*)d_in[5];
  const float* Wv = (const float*)d_in[6];
  const float* bv = (const float*)d_in[7];
  const float* gamma = (const float*)d_in[8];
  const float* beta  = (const float*)d_in[9];
  float* out = (float*)d_out;

  // Workspace layout (~88 MB).
  float* f_cur = (float*)d_ws;
  float* Qf    = f_cur + FSZ;
  float* Kf    = Qf + FSZ;
  float* Vf    = Kf + FSZ;
  float* xout  = Vf + FSZ;
  float* gsum  = xout + FSZ;         // 64 + 64 floats
  int*   next1 = (int*)(gsum + 128);
  int*   walk  = next1 + BN;

  hipMemcpyAsync(f_cur, feature, FSZ * sizeof(float),
                 hipMemcpyDeviceToDevice, stream);

  walk_top4_kernel<<<BN / 256, 256, 0, stream>>>(feature, knn, next1);
  walk_build_kernel<<<BN / 256, 256, 0, stream>>>(next1, walk);

  for (int L = 0; L < 3; ++L) {
    zero_kernel<<<1, 128, 0, stream>>>(gsum);
    dim3 g(256, BB);   // 2048 n-tiles / 8 waves per block, per batch
    gemm_qkv_kernel<<<g, 256, 0, stream>>>(
        f_cur,
        Wq + (size_t)L * CC * CC, bq + (size_t)L * CC,
        Wk + (size_t)L * CC * CC, bk + (size_t)L * CC,
        Wv + (size_t)L * CC * CC, bv + (size_t)L * CC,
        Qf, Kf, Vf);
    attn_kernel<<<BN / 4, 128, 0, stream>>>(Qf, Kf, Vf, walk, xout,
                                            gsum, gsum + 64);
    norm_kernel<<<(int)(FSZ / 256), 256, 0, stream>>>(
        xout, f_cur, gamma + (size_t)L * CC, beta + (size_t)L * CC,
        gsum, gsum + 64, out, L);
  }
}
